// Attention_11063835754934
// MI455X (gfx1250) — matrix operand order repaired and checked by reading, not compiled, not run on hardware
//
#include <hip/hip_runtime.h>

typedef _Float16 v16h __attribute__((ext_vector_type(16)));
typedef float    v8f  __attribute__((ext_vector_type(8)));

struct alignas(16) F4 { float x, y, z, w; };
struct H4 { _Float16 x, y, z, w; };
union  AB { v16h v; F4 f[2]; };

constexpr int kB   = 16;
constexpr int kNQ  = 2048;
constexpr int kNK  = 2048;
constexpr int kD   = 128;
constexpr int MBLK = 128;   // query rows per block (8 waves x 16)
constexpr int NT   = 32;    // keys per tile

__device__ __forceinline__ v8f wmma16(v16h a, v16h b, v8f c) {
  // D = A(16x32 f16) * B(32x16 f16) + C(16x16 f32)
  return __builtin_amdgcn_wmma_f32_16x16x32_f16(false, a, false, b, (short)0, c,
                                                false, false);
}

// A-fragment (16x32, f16) from row-major f16 storage.
// Lane: M = lane&15; hi half covers K = kb+8..15 / kb+24..31 (ISA 7.12.2).
__device__ __forceinline__ v16h load_a16(const _Float16* base, int strideH,
                                         int m, int kb, int off8) {
  AB a;
  const _Float16* p = base + m * strideH + kb + off8;
  a.f[0] = *(const F4*)p;         // K = kb+off8 .. +7
  a.f[1] = *(const F4*)(p + 16);  // K = kb+16+off8 .. +7
  return a.v;
}

// B-fragment (32x16, f16) from row-major [n][K] f16 storage.
// Lane: N = lane&15; hi half covers K = kb+8..15 / kb+24..31 (same as A).
__device__ __forceinline__ v16h load_b16(const _Float16* base, int strideH,
                                         int n, int kb, int off16) {
  AB b;
  const _Float16* p = base + n * strideH + kb + off16 / 2;
  b.f[0] = *(const F4*)p;
  b.f[1] = *(const F4*)(p + 16);
  return b.v;
}

// A-fragment built from an f32 global row (converted to f16).
__device__ __forceinline__ v16h load_a32(const float* row, int kb, int off8) {
  F4 a0 = *(const F4*)(row + kb + off8);
  F4 a1 = *(const F4*)(row + kb + off8 + 4);
  F4 b0 = *(const F4*)(row + kb + 16 + off8);
  F4 b1 = *(const F4*)(row + kb + 16 + off8 + 4);
  v16h r;
  r[0]=(_Float16)a0.x;  r[1]=(_Float16)a0.y;  r[2]=(_Float16)a0.z;  r[3]=(_Float16)a0.w;
  r[4]=(_Float16)a1.x;  r[5]=(_Float16)a1.y;  r[6]=(_Float16)a1.z;  r[7]=(_Float16)a1.w;
  r[8]=(_Float16)b0.x;  r[9]=(_Float16)b0.y;  r[10]=(_Float16)b0.z; r[11]=(_Float16)b0.w;
  r[12]=(_Float16)b1.x; r[13]=(_Float16)b1.y; r[14]=(_Float16)b1.z; r[15]=(_Float16)b1.w;
  return r;
}

// B-fragment built from f32 row-major global storage (converted to f16).
__device__ __forceinline__ v16h load_b32(const float* base, int strideF,
                                         int n, int kb, int off16) {
  const float* p = base + (size_t)n * strideF + kb + off16 / 2;
  F4 f0 = *(const F4*)p;
  F4 f1 = *(const F4*)(p + 4);
  F4 f2 = *(const F4*)(p + 16);
  F4 f3 = *(const F4*)(p + 20);
  v16h r;
  r[0]=(_Float16)f0.x;  r[1]=(_Float16)f0.y;  r[2]=(_Float16)f0.z;  r[3]=(_Float16)f0.w;
  r[4]=(_Float16)f1.x;  r[5]=(_Float16)f1.y;  r[6]=(_Float16)f1.z;  r[7]=(_Float16)f1.w;
  r[8]=(_Float16)f2.x;  r[9]=(_Float16)f2.y;  r[10]=(_Float16)f2.z; r[11]=(_Float16)f2.w;
  r[12]=(_Float16)f3.x; r[13]=(_Float16)f3.y; r[14]=(_Float16)f3.z; r[15]=(_Float16)f3.w;
  return r;
}

// Pull one 32-key K/V tile into registers (no LDS traffic yet).
__device__ __forceinline__ void load_tile_regs(const float* kg, const float* vg,
                                               int t, F4* kr, F4* vr) {
  #pragma unroll
  for (int j = 0; j < 4; ++j) {         // 1024 float4 / 256 threads
    int i   = t + j * 256;
    int key = i >> 5;                   // 32 float4 per 128-wide row
    int dc  = (i & 31) << 2;
    kr[j] = *(const F4*)(kg + key * kD + dc);
    vr[j] = *(const F4*)(vg + key * kD + dc);
  }
}

// Convert the register tile to f16 and commit to LDS (V transposed).
__device__ __forceinline__ void store_tile_lds(_Float16* hK, _Float16* hVt,
                                               int t, const F4* kr, const F4* vr) {
  #pragma unroll
  for (int j = 0; j < 4; ++j) {
    int i   = t + j * 256;
    int key = i >> 5;
    int dc  = (i & 31) << 2;
    H4 h;
    h.x = (_Float16)kr[j].x; h.y = (_Float16)kr[j].y;
    h.z = (_Float16)kr[j].z; h.w = (_Float16)kr[j].w;
    *(H4*)(hK + key * kD + dc) = h;
    hVt[(dc + 0) * NT + key] = (_Float16)vr[j].x;
    hVt[(dc + 1) * NT + key] = (_Float16)vr[j].y;
    hVt[(dc + 2) * NT + key] = (_Float16)vr[j].z;
    hVt[(dc + 3) * NT + key] = (_Float16)vr[j].w;
  }
}

__global__ void __launch_bounds__(256)
flash_attn_wmma(const float* __restrict__ q, const float* __restrict__ k,
                const float* __restrict__ v, const float* __restrict__ mask,
                const float* __restrict__ Wq, float* __restrict__ out)
{
  __shared__ __align__(16) _Float16 smem[16384];  // 32 KB (of 320 KB/WGP)
  _Float16* hQP = smem;            // phase 1: 128x128 projected queries (f16)
  _Float16* hK  = smem;            // phase 2: 32 keys x 128 (row-major)
  _Float16* hVt = smem + NT * kD;  // phase 2: 128 x 32 (V transposed)

  const int t    = threadIdx.x;
  const int w    = t >> 5;        // wave id (wave32)
  const int ln   = t & 31;
  const int lnm  = ln & 15;
  const int hi   = ln >> 4;
  const int off8  = hi ? 8 : 0;
  const int off16 = hi ? 16 : 0;
  const int b  = blockIdx.y;
  const int q0 = blockIdx.x * MBLK;
  _Float16* hP = smem + 2 * NT * kD + w * (16 * NT);  // per-wave P staging

  const v8f vzero = {0.f, 0.f, 0.f, 0.f, 0.f, 0.f, 0.f, 0.f};

  // ---------- Phase 1: query projection QP = Q @ Wq^T via WMMA ----------
  const float* qrow = q + ((size_t)b * kNQ + q0 + 16 * w + lnm) * kD;
  v16h aq[4];
  #pragma unroll
  for (int c = 0; c < 4; ++c) aq[c] = load_a32(qrow, 32 * c, off8);

  #pragma unroll
  for (int e = 0; e < 8; ++e) {                 // 8 output chunks of 16
    v8f acc = vzero;
    #pragma unroll
    for (int c = 0; c < 4; ++c) {               // K-dim 128 in chunks of 32
      v16h bw = load_b32(Wq, kD, e * 16 + lnm, 32 * c, off16);
      acc = wmma16(aq[c], bw, acc);
    }
    #pragma unroll
    for (int r = 0; r < 8; ++r)                 // C layout: row = r + 8*hi
      hQP[(16 * w + r + off8) * kD + e * 16 + lnm] = (_Float16)acc[r];
  }
  // Re-enter A-fragment layout (same wave wrote these rows; LDS is in-order).
  v16h qa[4];
  #pragma unroll
  for (int c = 0; c < 4; ++c)
    qa[c] = load_a16(hQP, kD, 16 * w + lnm, 32 * c, off8);

  // ---------- Phase 2: streaming (flash) softmax over key tiles ----------
  float m_run[8], l_run[8];
  v8f o[8];
  #pragma unroll
  for (int r = 0; r < 8; ++r) { m_run[r] = -1e30f; l_run[r] = 0.f; }
  #pragma unroll
  for (int d = 0; d < 8; ++d) o[d] = vzero;

  const float scale = 0.088388347648318447f;    // 1/sqrt(128)
  const float* mbase =
      mask + ((size_t)b * kNQ + q0 + 16 * w + off8) * kNK + lnm;
  const float* kbat = k + (size_t)b * kNK * kD;
  const float* vbat = v + (size_t)b * kNK * kD;

  // Prologue: tile 0 into registers.
  F4 kr[4], vr[4];
  load_tile_regs(kbat, vbat, t, kr, vr);

  for (int kt = 0; kt < kNK; kt += NT) {
    __syncthreads();                  // all waves done reading previous tile
    store_tile_lds(hK, hVt, t, kr, vr);
    __syncthreads();                  // tile kt visible to all waves

    // Kick off next tile's global loads; latency hides under this tile's work.
    if (kt + NT < kNK)
      load_tile_regs(kbat + (size_t)(kt + NT) * kD,
                     vbat + (size_t)(kt + NT) * kD, t, kr, vr);

    // Issue mask loads early (the roofline-critical stream), overlap WMMAs.
    const float* mrow = mbase + kt;
    __builtin_prefetch(mrow + NT, 0, 0);        // global_prefetch_b8 next tile
    float mk0[8], mk1[8];
    #pragma unroll
    for (int r = 0; r < 8; ++r) {
      mk0[r] = mrow[(size_t)r * kNK];
      mk1[r] = mrow[(size_t)r * kNK + 16];
    }

    // S(16x32) = QP_tile @ K_tile^T : 8 WMMAs
    v8f s0 = vzero, s1 = vzero;
    #pragma unroll
    for (int c = 0; c < 4; ++c) {
      s0 = wmma16(qa[c], load_b16(hK, kD, lnm,      32 * c, off16), s0);
      s1 = wmma16(qa[c], load_b16(hK, kD, 16 + lnm, 32 * c, off16), s1);
    }

    // Fused log(mask) + online softmax (rows live in 16-lane halves).
    #pragma unroll
    for (int r = 0; r < 8; ++r) {
      float x0 = s0[r] * scale + __logf(mk0[r]);
      float x1 = s1[r] * scale + __logf(mk1[r]);
      float mx = fmaxf(x0, x1);
      mx = fmaxf(mx, __shfl_xor(mx, 1, 32));
      mx = fmaxf(mx, __shfl_xor(mx, 2, 32));
      mx = fmaxf(mx, __shfl_xor(mx, 4, 32));
      mx = fmaxf(mx, __shfl_xor(mx, 8, 32));
      float mnew  = fmaxf(m_run[r], mx);
      float alpha = __expf(m_run[r] - mnew);
      m_run[r] = mnew;
      float p0 = __expf(x0 - mnew);
      float p1 = __expf(x1 - mnew);
      float rs = p0 + p1;
      rs += __shfl_xor(rs, 1, 32);
      rs += __shfl_xor(rs, 2, 32);
      rs += __shfl_xor(rs, 4, 32);
      rs += __shfl_xor(rs, 8, 32);
      l_run[r] = l_run[r] * alpha + rs;
      #pragma unroll
      for (int d = 0; d < 8; ++d) o[d][r] *= alpha;
      hP[(r + off8) * NT + lnm]      = (_Float16)p0;  // per-wave staging
      hP[(r + off8) * NT + lnm + 16] = (_Float16)p1;
    }

    // O(16x128) += P(16x32) @ V_tile(32x128) : 8 WMMAs
    v16h pa = load_a16(hP, NT, lnm, 0, off8);
    #pragma unroll
    for (int d = 0; d < 8; ++d)
      o[d] = wmma16(pa, load_b16(hVt, NT, d * 16 + lnm, 0, off16), o[d]);
  }

  // ---------- Epilogue: normalize and store f32 ----------
  float* orow = out + ((size_t)b * kNQ + q0 + 16 * w + off8) * kD + lnm;
  #pragma unroll
  for (int r = 0; r < 8; ++r) {
    float inv = 1.0f / l_run[r];
    #pragma unroll
    for (int d = 0; d < 8; ++d)
      orow[(size_t)r * kD + d * 16] = o[d][r] * inv;
  }
}

extern "C" void kernel_launch(void* const* d_in, const int* in_sizes, int n_in,
                              void* d_out, int out_size, void* d_ws, size_t ws_size,
                              hipStream_t stream) {
  (void)in_sizes; (void)n_in; (void)out_size; (void)d_ws; (void)ws_size;
  const float* q    = (const float*)d_in[0];
  const float* k    = (const float*)d_in[1];
  const float* v    = (const float*)d_in[2];
  const float* mask = (const float*)d_in[3];
  const float* Wq   = (const float*)d_in[4];
  float* out = (float*)d_out;
  dim3 grid(kNQ / MBLK, kB);   // 16 x 16 = 256 workgroups
  flash_attn_wmma<<<grid, dim3(256), 0, stream>>>(q, k, v, mask, Wq, out);
}
